// SparseLinear_31825707663797
// MI455X (gfx1250) — matrix operand order repaired and checked by reading, not compiled, not run on hardware
//
#include <hip/hip_runtime.h>
#include <hip/hip_bf16.h>

// out = relu(((din!=0) ? din+bias : 0) @ weight)   B=8192, D=4096, fp32 in/out.
// Compute-bound (AI ~820 FLOP/B) -> bf16 WMMA w/ f32 accumulate.
// Phase 1: convert din(+bias,mask) -> bf16 A' and weight -> bf16 W'^T in d_ws.
// Phase 2: GEMM staged by TDM tensor_load_to_lds (BK=64, double-buffered).
#define BATCH 8192
#define DIM   4096

typedef __attribute__((ext_vector_type(16))) __bf16 v16bf;
typedef __attribute__((ext_vector_type(8)))  __bf16 v8bf;
typedef __attribute__((ext_vector_type(2)))  __bf16 v2bf;
typedef __attribute__((ext_vector_type(8)))  float  v8f;
typedef unsigned int u32x4 __attribute__((ext_vector_type(4)));
typedef int          i32x8 __attribute__((ext_vector_type(8)));
typedef int          i32x4 __attribute__((ext_vector_type(4)));

// ---- pack two f32 -> bf16x2 dword. One v_add per value + one v_perm. ----
__device__ __forceinline__ unsigned int pk_bf16(float lo, float hi) {
#if __has_builtin(__builtin_amdgcn_cvt_pk_bf16_f32)
  v2bf p = __builtin_amdgcn_cvt_pk_bf16_f32(lo, hi);
  return __builtin_bit_cast(unsigned int, p);
#else
  unsigned int ua = __builtin_bit_cast(unsigned int, lo) + 0x8000u; // rnd-half-up
  unsigned int ub = __builtin_bit_cast(unsigned int, hi) + 0x8000u;
  return __builtin_amdgcn_perm(ub, ua, 0x07060302u); // {hi16(ub), hi16(ua)}
#endif
}

union AB16 { v16bf v; v8bf h[2]; };

// ---- 16 WMMAs on one staged K=64 chunk (per-wave 32x64 sub-tile) ----
// B fragments loaded one at a time to keep VGPR pressure low (acc stays put).
__device__ __forceinline__ void compute_tiles(
    const unsigned short (*As)[72], const unsigned short (*Bs)[72],
    int wm, int wn, int g, int l16, v8f acc[2][4]) {
  #pragma unroll
  for (int sub = 0; sub < 2; ++sub) {
    const int kb = sub * 32;
    AB16 a[2];
    #pragma unroll
    for (int t = 0; t < 2; ++t) {  // A 16x32: VGPR0-3 K=8g.., VGPR4-7 K=16+8g..
      const int m = wm * 32 + t * 16 + l16;
      a[t].h[0] = *(const v8bf*)&As[m][kb + g * 8];
      a[t].h[1] = *(const v8bf*)&As[m][kb + 16 + g * 8];
    }
    #pragma unroll
    for (int tn = 0; tn < 4; ++tn) { // B 32x16: VGPR0-3 K=8g.., VGPR4-7 K=16+8g..
      const int n = wn * 64 + tn * 16 + l16;
      AB16 b;
      b.h[0] = *(const v8bf*)&Bs[n][kb + g * 8];
      b.h[1] = *(const v8bf*)&Bs[n][kb + 16 + g * 8];
      acc[0][tn] = __builtin_amdgcn_wmma_f32_16x16x32_bf16(
          false, a[0].v, false, b.v, (short)0, acc[0][tn], false, false);
      acc[1][tn] = __builtin_amdgcn_wmma_f32_16x16x32_bf16(
          false, a[1].v, false, b.v, (short)0, acc[1][tn], false, false);
    }
  }
}

// ================= Phase 1 kernels =================

// A' = bf16( (din!=0) ? din+bias : 0 ), row-major [BATCH][DIM]. 8 elems/thread.
__global__ __launch_bounds__(256)
void convert_a(const float* __restrict__ din, const float* __restrict__ bias,
               unsigned short* __restrict__ Ab) {
  const size_t idx = ((size_t)blockIdx.x * 256 + threadIdx.x) * 8;
  const int col = (int)(idx & (DIM - 1));
  float4 v0 = *(const float4*)(din + idx);
  float4 v1 = *(const float4*)(din + idx + 4);
  float4 b0 = *(const float4*)(bias + col);
  float4 b1 = *(const float4*)(bias + col + 4);
  v0.x = (v0.x != 0.0f) ? v0.x + b0.x : 0.0f;
  v0.y = (v0.y != 0.0f) ? v0.y + b0.y : 0.0f;
  v0.z = (v0.z != 0.0f) ? v0.z + b0.z : 0.0f;
  v0.w = (v0.w != 0.0f) ? v0.w + b0.w : 0.0f;
  v1.x = (v1.x != 0.0f) ? v1.x + b1.x : 0.0f;
  v1.y = (v1.y != 0.0f) ? v1.y + b1.y : 0.0f;
  v1.z = (v1.z != 0.0f) ? v1.z + b1.z : 0.0f;
  v1.w = (v1.w != 0.0f) ? v1.w + b1.w : 0.0f;
  uint4 o;
  o.x = pk_bf16(v0.x, v0.y); o.y = pk_bf16(v0.z, v0.w);
  o.z = pk_bf16(v1.x, v1.y); o.w = pk_bf16(v1.z, v1.w);
  *(uint4*)(Ab + idx) = o;
}

// W'^T[n][k] = bf16(weight[k][n]) via 64x64 LDS tile transpose.
__global__ __launch_bounds__(256)
void convert_w(const float* __restrict__ w, unsigned short* __restrict__ Wt) {
  __shared__ float tw[64][65];
  const int n0 = blockIdx.x * 64, k0 = blockIdx.y * 64;
  const int r = threadIdx.x >> 2;          // 0..63
  const int c = (threadIdx.x & 3) * 16;    // 0,16,32,48
  #pragma unroll
  for (int j = 0; j < 4; ++j) {
    float4 v = *(const float4*)(w + (size_t)(k0 + r) * DIM + n0 + c + 4 * j);
    tw[r][c + 4 * j + 0] = v.x; tw[r][c + 4 * j + 1] = v.y;
    tw[r][c + 4 * j + 2] = v.z; tw[r][c + 4 * j + 3] = v.w;
  }
  __syncthreads();
  float f[16];
  #pragma unroll
  for (int i = 0; i < 16; ++i) f[i] = tw[c + i][r];   // transposed read
  uint4 o0, o1;
  o0.x = pk_bf16(f[0], f[1]);  o0.y = pk_bf16(f[2], f[3]);
  o0.z = pk_bf16(f[4], f[5]);  o0.w = pk_bf16(f[6], f[7]);
  o1.x = pk_bf16(f[8], f[9]);  o1.y = pk_bf16(f[10], f[11]);
  o1.z = pk_bf16(f[12], f[13]); o1.w = pk_bf16(f[14], f[15]);
  unsigned short* dst = Wt + (size_t)(n0 + r) * DIM + k0 + c;
  *(uint4*)(dst + 0) = o0;
  *(uint4*)(dst + 8) = o1;
}

// ================= Phase 2: GEMM =================

#if __has_builtin(__builtin_amdgcn_tensor_load_to_lds) && \
    __has_builtin(__builtin_amdgcn_s_wait_tensorcnt)
#define USE_TDM 1
// TDM load of a 128-row x 128-byte bf16 tile into LDS, padding each
// 32-dword row with 4 dwords -> 144 B LDS row stride (our [..][72] layout).
// 6-arg builtin form: (u32x4 g0, i32x8 g1, i32x4 g2, i32x4 g3, i32x8, i32 cpol)
__device__ __forceinline__ void tdm_load_tile(unsigned lds_byte_off,
                                              const void* gtile,
                                              unsigned tensor_rows) {
  const unsigned long long ga = (unsigned long long)(uintptr_t)gtile;
  u32x4 g0;
  g0[0] = 1u;                                    // count=1, user descriptor
  g0[1] = lds_byte_off;                          // lds_addr
  g0[2] = (unsigned)ga;                          // global_addr[31:0]
  g0[3] = (unsigned)((ga >> 32) & 0x1FFFFFFu)    // global_addr[56:32]
        | (2u << 30);                            // type = 2 ("image")
  i32x8 g1;
  g1[0] = (int)((1u << 16)     // data_size = 1 -> 2 bytes
        |       (1u << 20)     // pad_enable
        |       (4u << 22)     // pad_interval code 4 -> 32 dwords (128 B row)
        |       (3u << 25));   // pad_amount  code 3 -> 4 dwords  (16 B pad)
  g1[1] = (int)((DIM & 0xFFFFu) << 16);                        // dim0[15:0]
  g1[2] = (int)(((DIM >> 16) & 0xFFFFu) |
                ((tensor_rows & 0xFFFFu) << 16));              // dim0[31:16],dim1[15:0]
  g1[3] = (int)(((tensor_rows >> 16) & 0xFFFFu) | (64u << 16)); // dim1[31:16],tile_dim0=64
  g1[4] = (int)128u;                                           // tile_dim1=128 rows
  g1[5] = (int)DIM;                                            // dim0_stride (elems)
  g1[6] = 0; g1[7] = 0;                                        // dim1_stride unused (2D)
  i32x4 z4 = {0, 0, 0, 0};
  i32x8 z8 = {0, 0, 0, 0, 0, 0, 0, 0};
  __builtin_amdgcn_tensor_load_to_lds(g0, g1, z4, z4, z8, 0);
}
#endif

__global__ __launch_bounds__(256)
void gemm_bf16_fast(const unsigned short* __restrict__ Ab,
                    const unsigned short* __restrict__ Wt,
                    float* __restrict__ out) {
  __shared__ alignas(16) unsigned short As[2][128][72];
  __shared__ alignas(16) unsigned short Bs[2][128][72];

  const int tid  = threadIdx.x;
  const int lane = tid & 31;
  const int wid  = tid >> 5;
  const int wm   = wid >> 1;
  const int wn   = wid & 1;
  const int g    = lane >> 4;
  const int l16  = lane & 15;
  const int n0 = blockIdx.x * 128;
  const int m0 = blockIdx.y * 128;

  v8f acc[2][4];
  #pragma unroll
  for (int i = 0; i < 2; ++i)
    #pragma unroll
    for (int j = 0; j < 4; ++j) { v8f z = {}; acc[i][j] = z; }

#ifdef USE_TDM
  if (wid == 0) {
    tdm_load_tile((unsigned)(uintptr_t)&As[0][0][0], Ab + (size_t)m0 * DIM, BATCH);
    tdm_load_tile((unsigned)(uintptr_t)&Bs[0][0][0], Wt + (size_t)n0 * DIM, DIM);
    __builtin_amdgcn_s_wait_tensorcnt(0);
  }
  __syncthreads();
  int buf = 0;
  for (int kk = 64; kk < DIM; kk += 64) {
    if (wid == 0) {  // DMA next chunk while everyone computes on current
      tdm_load_tile((unsigned)(uintptr_t)&As[buf ^ 1][0][0],
                    Ab + (size_t)m0 * DIM + kk, BATCH);
      tdm_load_tile((unsigned)(uintptr_t)&Bs[buf ^ 1][0][0],
                    Wt + (size_t)n0 * DIM + kk, DIM);
    }
    compute_tiles(As[buf], Bs[buf], wm, wn, g, l16, acc);
    if (wid == 0) __builtin_amdgcn_s_wait_tensorcnt(0);
    __syncthreads();
    buf ^= 1;
  }
  compute_tiles(As[buf], Bs[buf], wm, wn, g, l16, acc);
#else
  // Fallback staging: pure b128 copy, register-pipelined double buffer.
  // 128 rows x 8 chunks(16B) per tile = 1024 chunks; 4 per thread.
  uint4 ra[4], rb[4];
  #pragma unroll
  for (int it = 0; it < 4; ++it) {
    const int c = tid + it * 256, row = c >> 3, q = c & 7;
    ra[it] = *(const uint4*)(Ab + (size_t)(m0 + row) * DIM + q * 8);
    rb[it] = *(const uint4*)(Wt + (size_t)(n0 + row) * DIM + q * 8);
  }
  #pragma unroll
  for (int it = 0; it < 4; ++it) {
    const int c = tid + it * 256, row = c >> 3, q = c & 7;
    *(uint4*)&As[0][row][q * 8] = ra[it];
    *(uint4*)&Bs[0][row][q * 8] = rb[it];
  }
  __syncthreads();
  int buf = 0;
  for (int kk = 64; kk < DIM; kk += 64) {
    #pragma unroll
    for (int it = 0; it < 4; ++it) {
      const int c = tid + it * 256, row = c >> 3, q = c & 7;
      ra[it] = *(const uint4*)(Ab + (size_t)(m0 + row) * DIM + kk + q * 8);
      rb[it] = *(const uint4*)(Wt + (size_t)(n0 + row) * DIM + kk + q * 8);
    }
    compute_tiles(As[buf], Bs[buf], wm, wn, g, l16, acc);
    #pragma unroll
    for (int it = 0; it < 4; ++it) {
      const int c = tid + it * 256, row = c >> 3, q = c & 7;
      *(uint4*)&As[buf ^ 1][row][q * 8] = ra[it];
      *(uint4*)&Bs[buf ^ 1][row][q * 8] = rb[it];
    }
    __syncthreads();
    buf ^= 1;
  }
  compute_tiles(As[buf], Bs[buf], wm, wn, g, l16, acc);
#endif

  // Epilogue: ReLU + store. C/D layout: VGPR r -> M = r + 8g, N = l16.
  #pragma unroll
  for (int tm = 0; tm < 2; ++tm) {
    const int row = m0 + wm * 32 + tm * 16 + g * 8;
    #pragma unroll
    for (int tn = 0; tn < 4; ++tn) {
      const int col = n0 + wn * 64 + tn * 16 + l16;
      #pragma unroll
      for (int r = 0; r < 8; ++r) {
        float v = acc[tm][tn][r];
        out[(size_t)(row + r) * DIM + col] = v > 0.0f ? v : 0.0f;
      }
    }
  }
}

// ============ Fallback all-in-one kernel (ws too small): BK=32 fused path ============

__device__ __forceinline__ void compute_tiles32(
    const unsigned short (*As)[40], const unsigned short (*Bs)[40],
    int wm, int wn, int g, int l16, v8f acc[2][4]) {
  AB16 a[2];
  #pragma unroll
  for (int t = 0; t < 2; ++t) {
    const int m = wm * 32 + t * 16 + l16;
    a[t].h[0] = *(const v8bf*)&As[m][g * 8];
    a[t].h[1] = *(const v8bf*)&As[m][16 + g * 8];
  }
  #pragma unroll
  for (int tn = 0; tn < 4; ++tn) {
    const int n = wn * 64 + tn * 16 + l16;
    AB16 b;
    b.h[0] = *(const v8bf*)&Bs[n][g * 8];
    b.h[1] = *(const v8bf*)&Bs[n][16 + g * 8];
    acc[0][tn] = __builtin_amdgcn_wmma_f32_16x16x32_bf16(
        false, a[0].v, false, b.v, (short)0, acc[0][tn], false, false);
    acc[1][tn] = __builtin_amdgcn_wmma_f32_16x16x32_bf16(
        false, a[1].v, false, b.v, (short)0, acc[1][tn], false, false);
  }
}

__device__ __forceinline__ void load_chunk(
    const float* __restrict__ din, const float* __restrict__ weight,
    const float* __restrict__ bias, int m0, int n0, int kk,
    int arow, int acol, int brow, int bn,
    float4 aR[4], float4& bvR, float4 bR[2][2]) {
  bvR = *(const float4*)(bias + kk + acol);
  #pragma unroll
  for (int i = 0; i < 4; ++i)
    aR[i] = *(const float4*)(din + (size_t)(m0 + arow + 32 * i) * DIM + kk + acol);
  #pragma unroll
  for (int wi = 0; wi < 2; ++wi) {
    const int k = 2 * (brow + 8 * wi);
    bR[wi][0] = *(const float4*)(weight + (size_t)(kk + k    ) * DIM + n0 + bn);
    bR[wi][1] = *(const float4*)(weight + (size_t)(kk + k + 1) * DIM + n0 + bn);
  }
}

__device__ __forceinline__ void store_chunk(
    unsigned short (*As)[40], unsigned short (*Bs)[40],
    int arow, int acol, int brow, int bn,
    const float4 aR[4], const float4& bvR, const float4 bR[2][2]) {
  #pragma unroll
  for (int i = 0; i < 4; ++i) {
    float4 v = aR[i];
    v.x = (v.x != 0.0f) ? v.x + bvR.x : 0.0f;
    v.y = (v.y != 0.0f) ? v.y + bvR.y : 0.0f;
    v.z = (v.z != 0.0f) ? v.z + bvR.z : 0.0f;
    v.w = (v.w != 0.0f) ? v.w + bvR.w : 0.0f;
    *(uint2*)&As[arow + 32 * i][acol] =
        make_uint2(pk_bf16(v.x, v.y), pk_bf16(v.z, v.w));
  }
  #pragma unroll
  for (int wi = 0; wi < 2; ++wi) {
    const int kp = brow + 8 * wi;
    const float* r0 = &bR[wi][0].x;
    const float* r1 = &bR[wi][1].x;
    #pragma unroll
    for (int j = 0; j < 4; ++j)
      *(unsigned int*)&Bs[bn + j][2 * kp] = pk_bf16(r0[j], r1[j]);
  }
}

__global__ __launch_bounds__(256)
void sparse_linear_fused(const float* __restrict__ din,
                         const float* __restrict__ weight,
                         const float* __restrict__ bias,
                         float* __restrict__ out) {
  __shared__ alignas(16) unsigned short As[2][128][40];
  __shared__ alignas(16) unsigned short Bs[2][128][40];
  const int tid = threadIdx.x, lane = tid & 31, wid = tid >> 5;
  const int wm = wid >> 1, wn = wid & 1, g = lane >> 4, l16 = lane & 15;
  const int n0 = blockIdx.x * 128, m0 = blockIdx.y * 128;
  const int acol = (tid & 7) * 4, arow = tid >> 3;
  const int bn = (tid & 31) * 4, brow = tid >> 5;

  v8f acc[2][4];
  #pragma unroll
  for (int i = 0; i < 2; ++i)
    #pragma unroll
    for (int j = 0; j < 4; ++j) { v8f z = {}; acc[i][j] = z; }

  float4 aR[4], bR[2][2], bvR;
  load_chunk(din, weight, bias, m0, n0, 0, arow, acol, brow, bn, aR, bvR, bR);
  store_chunk(As[0], Bs[0], arow, acol, brow, bn, aR, bvR, bR);
  __syncthreads();
  int buf = 0;
  for (int kk = 32; kk < DIM; kk += 32) {
    load_chunk(din, weight, bias, m0, n0, kk, arow, acol, brow, bn, aR, bvR, bR);
    compute_tiles32(As[buf], Bs[buf], wm, wn, g, l16, acc);
    store_chunk(As[buf ^ 1], Bs[buf ^ 1], arow, acol, brow, bn, aR, bvR, bR);
    __syncthreads();
    buf ^= 1;
  }
  compute_tiles32(As[buf], Bs[buf], wm, wn, g, l16, acc);

  #pragma unroll
  for (int tm = 0; tm < 2; ++tm) {
    const int row = m0 + wm * 32 + tm * 16 + g * 8;
    #pragma unroll
    for (int tn = 0; tn < 4; ++tn) {
      const int col = n0 + wn * 64 + tn * 16 + l16;
      #pragma unroll
      for (int r = 0; r < 8; ++r) {
        float v = acc[tm][tn][r];
        out[(size_t)(row + r) * DIM + col] = v > 0.0f ? v : 0.0f;
      }
    }
  }
}

extern "C" void kernel_launch(void* const* d_in, const int* in_sizes, int n_in,
                              void* d_out, int out_size, void* d_ws, size_t ws_size,
                              hipStream_t stream) {
  const float* din    = (const float*)d_in[0];
  const float* weight = (const float*)d_in[1];
  const float* bias   = (const float*)d_in[2];
  float*       out    = (float*)d_out;

  const size_t needA = (size_t)BATCH * DIM * sizeof(unsigned short);
  const size_t needW = (size_t)DIM * DIM * sizeof(unsigned short);
  if (ws_size >= needA + needW) {
    unsigned short* Ab = (unsigned short*)d_ws;
    unsigned short* Wt = Ab + (size_t)BATCH * DIM;
    convert_a<<<dim3((BATCH * (size_t)DIM) / 2048), 256, 0, stream>>>(din, bias, Ab);
    convert_w<<<dim3(DIM / 64, DIM / 64), 256, 0, stream>>>(weight, Wt);
    gemm_bf16_fast<<<dim3(DIM / 128, BATCH / 128), 256, 0, stream>>>(Ab, Wt, out);
  } else {
    sparse_linear_fused<<<dim3(DIM / 128, BATCH / 128), 256, 0, stream>>>(
        din, weight, bias, out);
  }
}
